// GNNModelWithResidual_163208757334
// MI455X (gfx1250) — matrix operand order repaired and checked by reading, not compiled, not run on hardware
//
#include <hip/hip_runtime.h>
#include <hip/hip_bf16.h>

// ---------------- problem constants (match reference) ----------------
#define NN    50000
#define EE    800000
#define ETOT  (EE + NN)      // edges + self loops (GAT)
#define DH    128
#define NHEAD 4
#define CH    32             // per-head channels
#define GG    50
#define DOUT  64
#define BNEPS 1e-5f

typedef __attribute__((ext_vector_type(16))) _Float16 v16h;
typedef __attribute__((ext_vector_type(8)))  float    v8f;

// =====================================================================
// WMMA GEMM:  Out[M,Ncol] = A[M,K] * W[K,Ncol] (+bias) (+Cin seed)
// Wave32; each wave computes a 64x16 macrotile (4 stacked 16x16 WMMA
// tiles) so the B fragment is loaded once per K-step and reused 4x.
// K is compile-time (=128 here) so the loop fully unrolls: 16 v_wmma.
// Row handling: loads are CLAMPED to M-1 (no divergent predication),
// stores are row-guarded. A and B both use contiguous k = half*16 + e.
// =====================================================================
template <int K>
__global__ void wmma_gemm_kernel(const float* __restrict__ A,
                                 const float* __restrict__ W,
                                 const float* __restrict__ bias,
                                 const float* __restrict__ Cin,
                                 float* __restrict__ Out,
                                 int M, int Ncol) {
  const int lane = threadIdx.x;      // 0..31
  const int half = lane >> 4;        // 0 | 1
  const int l16  = lane & 15;
  const int row0 = blockIdx.x * 64;
  const int col0 = (blockIdx.y * 4 + threadIdx.y) * 16;  // wave-uniform

  v8f acc[4];
#pragma unroll
  for (int t = 0; t < 4; ++t) {
    if (Cin != nullptr) {
#pragma unroll
      for (int v = 0; v < 8; ++v) {
        int m = row0 + t * 16 + v + 8 * half;
        int mc = (m < M) ? m : (M - 1);
        acc[t][v] = Cin[(size_t)mc * Ncol + col0 + l16];
      }
    } else {
#pragma unroll
      for (int v = 0; v < 8; ++v) acc[t][v] = 0.0f;
    }
  }

#pragma unroll
  for (int kt = 0; kt < K; kt += 32) {
    // ---- B fragment: 32x16 f16, col = l16, k = half*16 + e ----
    v16h b;
#pragma unroll
    for (int e = 0; e < 16; ++e) {
      int k = half * 16 + e;
      b[e] = (_Float16)W[(size_t)(kt + k) * Ncol + col0 + l16];
    }
    // ---- 4 stacked A fragments, each 16x32 f16 ----
#pragma unroll
    for (int t = 0; t < 4; ++t) {
      int arow = row0 + t * 16 + l16;
      arow = (arow < M) ? arow : (M - 1);           // clamp: no divergence
      const float* ap = A + (size_t)arow * K + kt + half * 16;
      float4 a0 = *(const float4*)(ap);
      float4 a1 = *(const float4*)(ap + 4);
      float4 a2 = *(const float4*)(ap + 8);
      float4 a3 = *(const float4*)(ap + 12);
      v16h a;
      a[0] = (_Float16)a0.x;  a[1] = (_Float16)a0.y;
      a[2] = (_Float16)a0.z;  a[3] = (_Float16)a0.w;
      a[4] = (_Float16)a1.x;  a[5] = (_Float16)a1.y;
      a[6] = (_Float16)a1.z;  a[7] = (_Float16)a1.w;
      a[8] = (_Float16)a2.x;  a[9] = (_Float16)a2.y;
      a[10] = (_Float16)a2.z; a[11] = (_Float16)a2.w;
      a[12] = (_Float16)a3.x; a[13] = (_Float16)a3.y;
      a[14] = (_Float16)a3.z; a[15] = (_Float16)a3.w;
      acc[t] = __builtin_amdgcn_wmma_f32_16x16x32_f16(false, a, false, b,
                                                      (short)0, acc[t],
                                                      false, false);
    }
  }

  const float bv = (bias != nullptr) ? bias[col0 + l16] : 0.0f;
#pragma unroll
  for (int t = 0; t < 4; ++t) {
#pragma unroll
    for (int v = 0; v < 8; ++v) {
      int m = row0 + t * 16 + v + 8 * half;
      if (m < M) Out[(size_t)m * Ncol + col0 + l16] = acc[t][v] + bv;
    }
  }
}

// ---------------- elementwise / scatter kernels ----------------------
__global__ void fill_zero_kernel(float* p, int n) {
  int i = blockIdx.x * blockDim.x + threadIdx.x;
  if (i < n) p[i] = 0.0f;
}

// out[n,f] = bias[f]  (seeds GAT accumulator with the layer bias)
__global__ void init_bias_kernel(float* out, const float* __restrict__ bias, int M) {
  int idx = blockIdx.x * blockDim.x + threadIdx.x;
  if (idx < M * DH) out[idx] = bias[idx & (DH - 1)];
}

// es[n,h] = <h[n,h,:], a_src[h,:]> ; ed likewise
__global__ void attn_coef_kernel(const float* __restrict__ h,
                                 const float* __restrict__ as,
                                 const float* __restrict__ ad,
                                 float* __restrict__ es, float* __restrict__ ed) {
  int idx = blockIdx.x * blockDim.x + threadIdx.x;
  if (idx >= NN * NHEAD) return;
  int n = idx / NHEAD, hh = idx % NHEAD;
  const float* hp = h + (size_t)n * DH + hh * CH;
  const float* ap = as + hh * CH;
  const float* dp = ad + hh * CH;
  float s = 0.f, d = 0.f;
#pragma unroll
  for (int c = 0; c < CH; ++c) { s += hp[c] * ap[c]; d += hp[c] * dp[c]; }
  es[idx] = s; ed[idx] = d;
}

// per (edge, head): w = exp(lrelu(es[src]+ed[dst])); z[dst] += w
// (max-shift omitted: softmax is shift-invariant and every dst has a self loop)
__global__ void edge_pass1_kernel(const int* __restrict__ src, const int* __restrict__ dst,
                                  const float* __restrict__ es, const float* __restrict__ ed,
                                  float* __restrict__ w, float* __restrict__ z) {
  int idx = blockIdx.x * blockDim.x + threadIdx.x;
  if (idx >= ETOT * NHEAD) return;
  int e = idx / NHEAD, hh = idx % NHEAD;
  int s = (e < EE) ? src[e] : (e - EE);
  int d = (e < EE) ? dst[e] : (e - EE);
  float v = es[s * NHEAD + hh] + ed[d * NHEAD + hh];
  v = (v > 0.f) ? v : 0.2f * v;
  float wv = expf(v);
  w[idx] = wv;
  atomicAdd(&z[d * NHEAD + hh], wv);
}

// per (edge, feature): out[dst,f] += h[src,f] * w/z
__global__ void edge_pass2_kernel(const int* __restrict__ src, const int* __restrict__ dst,
                                  const float* __restrict__ hmat,
                                  const float* __restrict__ w, const float* __restrict__ z,
                                  float* __restrict__ out) {
  int idx = blockIdx.x * blockDim.x + threadIdx.x;
  if (idx >= ETOT * DH) return;  // 108.8M < 2^31
  int e = idx >> 7, f = idx & (DH - 1);
  int hh = f >> 5;
  int s = (e < EE) ? src[e] : (e - EE);
  int d = (e < EE) ? dst[e] : (e - EE);
  float alpha = w[e * NHEAD + hh] / fmaxf(z[d * NHEAD + hh], 1e-16f);
  atomicAdd(&out[(size_t)d * DH + f], hmat[(size_t)s * DH + f] * alpha);
}

// training-mode BN statistics: one block per column
__global__ void bn_stats_kernel(const float* __restrict__ x, float* __restrict__ stats, int M) {
  __shared__ float ssum[256], ssq[256];
  int d = blockIdx.x;
  float s = 0.f, q = 0.f;
  for (int n = threadIdx.x; n < M; n += blockDim.x) {
    float v = x[(size_t)n * DH + d];
    s += v; q += v * v;
  }
  ssum[threadIdx.x] = s; ssq[threadIdx.x] = q;
  __syncthreads();
  for (int off = 128; off > 0; off >>= 1) {
    if (threadIdx.x < off) {
      ssum[threadIdx.x] += ssum[threadIdx.x + off];
      ssq[threadIdx.x]  += ssq[threadIdx.x + off];
    }
    __syncthreads();
  }
  if (threadIdx.x == 0) {
    float mu  = ssum[0] / (float)M;
    float var = ssq[0] / (float)M - mu * mu;
    stats[d]      = mu;
    stats[DH + d] = rsqrtf(var + BNEPS);
  }
}

// y = lrelu( g*(x-mu)*rstd + b (+res) )
__global__ void bn_apply_kernel(const float* __restrict__ x, const float* __restrict__ stats,
                                const float* __restrict__ g, const float* __restrict__ b,
                                const float* __restrict__ res, float* __restrict__ out) {
  int idx = blockIdx.x * blockDim.x + threadIdx.x;
  if (idx >= NN * DH) return;
  int f = idx & (DH - 1);
  float v = g[f] * (x[idx] - stats[f]) * stats[DH + f] + b[f];
  if (res != nullptr) v += res[idx];
  out[idx] = (v > 0.f) ? v : 0.2f * v;
}

__global__ void deg_kernel(const int* __restrict__ dst, float* __restrict__ deg) {
  int e = blockIdx.x * blockDim.x + threadIdx.x;
  if (e < EE) atomicAdd(&deg[dst[e]], 1.0f);
}

__global__ void sage_scatter_kernel(const int* __restrict__ src, const int* __restrict__ dst,
                                    const float* __restrict__ x, float* __restrict__ acc) {
  int idx = blockIdx.x * blockDim.x + threadIdx.x;
  if (idx >= EE * DH) return;  // 102.4M < 2^31
  int e = idx >> 7, f = idx & (DH - 1);
  atomicAdd(&acc[(size_t)dst[e] * DH + f], x[(size_t)src[e] * DH + f]);
}

__global__ void sage_divide_kernel(float* __restrict__ acc, const float* __restrict__ deg) {
  int idx = blockIdx.x * blockDim.x + threadIdx.x;
  if (idx >= NN * DH) return;
  acc[idx] /= fmaxf(deg[idx >> 7], 1.0f);
}

__global__ void pool_scatter_kernel(const int* __restrict__ batch, const float* __restrict__ x,
                                    float* __restrict__ pooled, float* __restrict__ cnt) {
  int idx = blockIdx.x * blockDim.x + threadIdx.x;
  if (idx >= NN * DH) return;
  int n = idx >> 7, f = idx & (DH - 1);
  int g = batch[n];
  atomicAdd(&pooled[(size_t)g * DH + f], x[idx]);
  if (f == 0) atomicAdd(&cnt[g], 1.0f);
}

__global__ void pool_div_kernel(float* __restrict__ pooled, const float* __restrict__ cnt) {
  int idx = blockIdx.x * blockDim.x + threadIdx.x;
  if (idx >= GG * DH) return;
  pooled[idx] /= fmaxf(cnt[idx >> 7], 1.0f);
}

// =====================================================================
extern "C" void kernel_launch(void* const* d_in, const int* in_sizes, int n_in,
                              void* d_out, int out_size, void* d_ws, size_t ws_size,
                              hipStream_t stream) {
  // ---- inputs (setup_inputs dict order) ----
  const float* x       = (const float*)d_in[0];
  const int*   ei      = (const int*)  d_in[1];
  const int*   batch   = (const int*)  d_in[2];
  const float* gat1_W  = (const float*)d_in[3];
  const float* gat1_as = (const float*)d_in[4];
  const float* gat1_ad = (const float*)d_in[5];
  const float* gat1_b  = (const float*)d_in[6];
  const float* gat2_W  = (const float*)d_in[7];
  const float* gat2_as = (const float*)d_in[8];
  const float* gat2_ad = (const float*)d_in[9];
  const float* gat2_b  = (const float*)d_in[10];
  const float* sWl[3]  = {(const float*)d_in[11], (const float*)d_in[14], (const float*)d_in[17]};
  const float* sbl[3]  = {(const float*)d_in[12], (const float*)d_in[15], (const float*)d_in[18]};
  const float* sWr[3]  = {(const float*)d_in[13], (const float*)d_in[16], (const float*)d_in[19]};
  const float* bng[5]  = {(const float*)d_in[20], (const float*)d_in[22], (const float*)d_in[24],
                          (const float*)d_in[26], (const float*)d_in[28]};
  const float* bnb[5]  = {(const float*)d_in[21], (const float*)d_in[23], (const float*)d_in[25],
                          (const float*)d_in[27], (const float*)d_in[29]};
  const float* res1_W  = (const float*)d_in[30];
  const float* res1_b  = (const float*)d_in[31];
  const float* res2_W  = (const float*)d_in[32];
  const float* res2_b  = (const float*)d_in[33];
  const float* fc_W    = (const float*)d_in[34];
  const float* fc_b    = (const float*)d_in[35];

  const int* src = ei;        // edge_index[0]
  const int* dst = ei + EE;   // edge_index[1]

  // ---- workspace carve-up (floats) ----
  float* ws = (float*)d_ws;
  const size_t ND = (size_t)NN * DH;
  float* Abuf  = ws;                 // activations
  float* Rbuf  = Abuf  + ND;         // residual
  float* Hbuf  = Rbuf  + ND;         // GAT-transformed h / SAGE mean
  float* Obuf  = Hbuf  + ND;         // atomic accumulation output
  float* wEdge = Obuf  + ND;         // ETOT*NHEAD
  float* zBuf  = wEdge + (size_t)ETOT * NHEAD;  // NN*NHEAD
  float* esBuf = zBuf  + (size_t)NN * NHEAD;
  float* edBuf = esBuf + (size_t)NN * NHEAD;
  float* degB  = edBuf + (size_t)NN * NHEAD;    // NN
  float* stats = degB  + NN;                    // 2*DH
  float* pooled= stats + 2 * DH;                // GG*DH
  float* cntB  = pooled + (size_t)GG * DH;      // GG

  const int TB = 256;
  auto blocks = [](int n, int tb) { return (n + tb - 1) / tb; };

  dim3 gB(32, 4);
  auto gemm = [&](const float* A, const float* W, const float* bias,
                  const float* Cin, float* Out, int M, int Ncol) {
    dim3 grid((M + 63) / 64, (Ncol + 63) / 64);
    wmma_gemm_kernel<DH><<<grid, gB, 0, stream>>>(A, W, bias, Cin, Out, M, Ncol);
  };
  auto gat = [&](const float* xin, const float* W, const float* as, const float* ad,
                 const float* b, const float* g_, const float* bb,
                 const float* res, float* hout) {
    gemm(xin, W, nullptr, nullptr, Hbuf, NN, DH);
    attn_coef_kernel<<<blocks(NN * NHEAD, TB), TB, 0, stream>>>(Hbuf, as, ad, esBuf, edBuf);
    fill_zero_kernel<<<blocks(NN * NHEAD, TB), TB, 0, stream>>>(zBuf, NN * NHEAD);
    init_bias_kernel<<<blocks(NN * DH, TB), TB, 0, stream>>>(Obuf, b, NN);
    edge_pass1_kernel<<<blocks(ETOT * NHEAD, TB), TB, 0, stream>>>(src, dst, esBuf, edBuf, wEdge, zBuf);
    edge_pass2_kernel<<<blocks(ETOT * DH, TB), TB, 0, stream>>>(src, dst, Hbuf, wEdge, zBuf, Obuf);
    bn_stats_kernel<<<DH, 256, 0, stream>>>(Obuf, stats, NN);
    bn_apply_kernel<<<blocks(NN * DH, TB), TB, 0, stream>>>(Obuf, stats, g_, bb, res, hout);
  };

  // -------- layer 1: res1 + GAT1 + BN1 + residual + lrelu --------
  gemm(x, res1_W, res1_b, nullptr, Rbuf, NN, DH);
  gat(x, gat1_W, gat1_as, gat1_ad, gat1_b, bng[0], bnb[0], Rbuf, Abuf);

  // -------- layer 2: res2 + GAT2 + BN2 + residual + lrelu --------
  gemm(Abuf, res2_W, res2_b, nullptr, Rbuf, NN, DH);
  gat(Abuf, gat2_W, gat2_as, gat2_ad, gat2_b, bng[1], bnb[1], Rbuf, Abuf);

  // -------- SAGE layers 3..5 (shared degree) --------
  fill_zero_kernel<<<blocks(NN, TB), TB, 0, stream>>>(degB, NN);
  deg_kernel<<<blocks(EE, TB), TB, 0, stream>>>(dst, degB);
  for (int i = 0; i < 3; ++i) {
    fill_zero_kernel<<<blocks(NN * DH, TB), TB, 0, stream>>>(Hbuf, NN * DH);
    sage_scatter_kernel<<<blocks(EE * DH, TB), TB, 0, stream>>>(src, dst, Abuf, Hbuf);
    sage_divide_kernel<<<blocks(NN * DH, TB), TB, 0, stream>>>(Hbuf, degB);
    gemm(Hbuf, sWl[i], sbl[i], nullptr, Obuf, NN, DH);      // mean @ Wl + bl
    gemm(Abuf, sWr[i], nullptr, Obuf, Obuf, NN, DH);        // += x @ Wr (seeded acc)
    bn_stats_kernel<<<DH, 256, 0, stream>>>(Obuf, stats, NN);
    bn_apply_kernel<<<blocks(NN * DH, TB), TB, 0, stream>>>(Obuf, stats, bng[2 + i], bnb[2 + i],
                                                            nullptr, Abuf);
  }

  // -------- global mean pool + fc --------
  fill_zero_kernel<<<blocks(GG * DH, TB), TB, 0, stream>>>(pooled, GG * DH);
  fill_zero_kernel<<<1, TB, 0, stream>>>(cntB, GG);
  pool_scatter_kernel<<<blocks(NN * DH, TB), TB, 0, stream>>>(batch, Abuf, pooled, cntB);
  pool_div_kernel<<<blocks(GG * DH, TB), TB, 0, stream>>>(pooled, cntB);
  gemm(pooled, fc_W, fc_b, nullptr, (float*)d_out, GG, DOUT);
}
